// FactorImpactAttention_55490977465200
// MI455X (gfx1250) — matrix operand order repaired and checked by reading, not compiled, not run on hardware
//
#include <hip/hip_runtime.h>
#include <hip/hip_bf16.h>
#include <math.h>

// ---------------- CDNA5 WMMA helpers (gfx1250, wave32) ----------------
typedef __bf16 bf16_t;
typedef __attribute__((ext_vector_type(16))) __bf16 v16bf;
typedef __attribute__((ext_vector_type(8)))  float  v8f;
typedef __attribute__((ext_vector_type(4)))  unsigned int v4u;

union ABFrag { v16bf bf; v4u q[2]; };

// A-matrix fragment (16x32 bf16, row-major source, tile origin pointer).
// ISA layout: lanes 0-15 row M=lane, K={0..7,16..23}; lanes 16-31 row M=lane-16, K={8..15,24..31}.
__device__ inline v16bf load_a_frag(const bf16_t* a, int lda, int lane) {
  const bf16_t* p = a + (lane & 15) * lda + ((lane >> 4) << 3);
  ABFrag f;
  f.q[0] = *(const v4u*)(p);        // K base..base+7
  f.q[1] = *(const v4u*)(p + 16);   // K base+16..base+23
  return f.bf;
}

// B-matrix fragment (32x16 bf16). B = W.T with W row-major (N x K):
// lane n<16 -> column n = W row n, K={0..7,16..23}; lanes 16-31 -> K={8..15,24..31}.
__device__ inline v16bf load_b_frag(const bf16_t* w, int ldw, int lane) {
  const bf16_t* p = w + (lane & 15) * ldw + ((lane >> 4) << 3);
  ABFrag f;
  f.q[0] = *(const v4u*)(p);
  f.q[1] = *(const v4u*)(p + 16);
  return f.bf;
}

// C/D f32 16x16: VGPR r -> row r (lanes<16) / r+8 (lanes>=16), col = lane&15.
__device__ inline void store_d_tile(float* c, int ldc, v8f d, int lane) {
  float* p = c + ((lane >> 4) << 3) * ldc + (lane & 15);
#pragma unroll
  for (int r = 0; r < 8; ++r) p[r * ldc] = d[r];
}

__device__ inline v8f wmma_bf16(v16bf a, v16bf b, v8f c) {
  return __builtin_amdgcn_wmma_f32_16x16x32_bf16(false, a, false, b, (short)0, c, false, false);
}

// gfx1250 hardware transcendental tanh (TRANS32). Trailing v_nop covers the
// documented TRANS co-execution hazard (1 indep op / V_NOP before use).
__device__ inline float fast_tanh(float x) {
  float r;
  asm volatile("v_tanh_f32 %0, %1\n\tv_nop" : "=v"(r) : "v"(x));
  return r;
}
__device__ inline float sigmoidf_(float x) { return 0.5f * fast_tanh(0.5f * x) + 0.5f; }

// Async global->LDS copy (16B per lane), tracked by ASYNCcnt.
__device__ inline void async_g2l_b128(unsigned lds_byte_off, const void* gptr) {
  asm volatile("global_load_async_to_lds_b128 %0, %1, off"
               :
               : "v"(lds_byte_off), "v"(gptr)
               : "memory");
}
__device__ inline void wait_async0() {
  asm volatile("s_wait_asynccnt 0x0" ::: "memory");
}
__device__ inline unsigned lds_off_of(const void* p) {
  return (unsigned)(uintptr_t)p;  // generic LDS addr: low 32 bits = LDS byte offset
}

// ---------------- small prep kernels ----------------
__global__ void f2bf_kernel(bf16_t* dst, const float* src, int n) {
  int i = blockIdx.x * 256 + threadIdx.x;
  if (i < n) dst[i] = (bf16_t)src[i];
}
__global__ void addv_kernel(float* dst, const float* a, const float* b, int n) {
  int i = blockIdx.x * 256 + threadIdx.x;
  if (i < n) dst[i] = a[i] + b[i];
}
__global__ void rowsum23_kernel(float* dst, const float* w, int rows) {
  int i = blockIdx.x * 256 + threadIdx.x;
  if (i < rows) {
    float s = 0.f;
    for (int f = 0; f < 23; ++f) s += w[i * 23 + f];
    dst[i] = s;
  }
}
__global__ void means_kernel(float* m, const float* factors, int n) {
  int i = blockIdx.x * 256 + threadIdx.x;
  if (i < n) {
    float s = 0.f;
    for (int f = 0; f < 23; ++f) s += factors[i * 23 + f];
    m[i] = s * (1.0f / 23.0f);
  }
}
__global__ void zero_kernel(float* p, int n) {
  int i = blockIdx.x * 256 + threadIdx.x;
  if (i < n) p[i] = 0.f;
}

// ---------------- encoder + attention + impacts (one block per batch elem) ----------------
__global__ __launch_bounds__(256) void attn_kernel(
    const float* __restrict__ factors,
    const float* __restrict__ few1, const float* __restrict__ feb1,
    const float* __restrict__ few2, const float* __restrict__ feb2,
    const float* __restrict__ few3, const float* __restrict__ feb3,
    const bf16_t* __restrict__ wqkv, const float* __restrict__ bqkv,
    const bf16_t* __restrict__ wout, const float* __restrict__ bout,
    const float* __restrict__ iw1, const float* __restrict__ ib1,
    const float* __restrict__ iw2, const float* __restrict__ ib2,
    float* __restrict__ out_impacts, float* __restrict__ out_attnavg) {
  // 64 KB LDS, phase-overlapped:
  __shared__ __align__(32) char smem[65536];
  float*  bigf  = (float*)smem;                 // 12288 f32: h1e/h2e -> qkv(32x384) -> attm(32x128)+ihs(9x256)
  bf16_t* xab   = (bf16_t*)(smem + 49152);      // 32x128 bf16 (then aavg/gvec/lgts region)
  bf16_t* oab   = (bf16_t*)(smem + 57344);      // 32x128 bf16 (first holds flast)
  float*  h1e   = bigf;                         // 23x32
  float*  h2e   = bigf + 768;                   // 23x64
  float*  qkv   = bigf;                         // 32x384
  float*  attm  = bigf;                         // 32x128
  float*  ihs   = bigf + 4096;                  // 9x256
  float*  aavg  = (float*)(smem + 49152);       // 529 (reuses dead xab)
  float*  gvec  = aavg + 544;                   // 128
  float*  lgts  = aavg + 680;                   // 207
  float*  flast = (float*)(smem + 57344);       // 23 (dead before oab used)

  const int t = threadIdx.x;
  const int lane = t & 31, wv = t >> 5;
  const int b = blockIdx.x;

  for (int i = t; i < 32 * 128; i += 256) xab[i] = (bf16_t)0.0f;  // zero incl. pad rows 23..31
  if (t < 23) flast[t] = factors[(b * 40 + 39) * 23 + t];
  __syncthreads();

  // factor-encoder MLP at last step only: 1 -> 32 -> 64 -> 128
  for (int i = t; i < 23 * 32; i += 256) {
    int j = i >> 5, u = i & 31;
    float v = flast[j] * few1[u] + feb1[u];
    h1e[i] = v > 0.f ? v : 0.f;
  }
  __syncthreads();
  for (int i = t; i < 23 * 64; i += 256) {
    int j = i >> 6, u = i & 63;
    const float* wr = few2 + u * 32;
    const float* hr = h1e + j * 32;
    float s = feb2[u];
    for (int k = 0; k < 32; ++k) s += hr[k] * wr[k];
    h2e[i] = s > 0.f ? s : 0.f;
  }
  __syncthreads();
  for (int i = t; i < 23 * 128; i += 256) {
    int j = i >> 7, u = i & 127;
    const float* wr = few3 + u * 64;
    const float* hr = h2e + j * 64;
    float s = feb3[u];
    for (int k = 0; k < 64; ++k) s += hr[k] * wr[k];
    xab[j * 128 + u] = (bf16_t)s;
  }
  __syncthreads();

  // QKV = x @ in_proj_w.T : (32pad x 128) @ (128 x 384) via WMMA bf16
  for (int tid = wv; tid < 48; tid += 8) {
    int mt = tid / 24, nt = tid % 24;
    v8f acc = {};
#pragma unroll
    for (int k0 = 0; k0 < 128; k0 += 32) {
      v16bf a  = load_a_frag(xab + mt * 16 * 128 + k0, 128, lane);
      v16bf bb = load_b_frag(wqkv + nt * 16 * 128 + k0, 128, lane);
      acc = wmma_bf16(a, bb, acc);
    }
    store_d_tile(qkv + mt * 16 * 384 + nt * 16, 384, acc, lane);
  }
  __syncthreads();

  for (int i = t; i < 23 * 384; i += 256) { int j = i / 384, n = i % 384; qkv[j * 384 + n] += bqkv[n]; }
  for (int i = t; i < 529; i += 256) aavg[i] = 0.f;
  for (int i = t; i < 32 * 128; i += 256) oab[i] = (bf16_t)0.0f;
  __syncthreads();

  // attention: 8 heads x 23 rows = 184 (head,row) pairs, VALU
  if (t < 184) {
    int hd = t / 23, r = t % 23;
    const float* qr = qkv + r * 384 + hd * 16;
    float sc[23];
    float mx = -1e30f;
    for (int j = 0; j < 23; ++j) {
      const float* kr = qkv + j * 384 + 128 + hd * 16;
      float s = 0.f;
      for (int c = 0; c < 16; ++c) s += qr[c] * kr[c];
      s *= 0.25f;  // 1/sqrt(16)
      sc[j] = s;
      mx = s > mx ? s : mx;
    }
    float den = 0.f;
    for (int j = 0; j < 23; ++j) { sc[j] = __expf(sc[j] - mx); den += sc[j]; }
    float inv = 1.f / den;
    for (int j = 0; j < 23; ++j) {
      sc[j] *= inv;
      atomicAdd(&aavg[r * 23 + j], sc[j] * 0.125f);  // mean over 8 heads
    }
    for (int c = 0; c < 16; ++c) {
      float o = 0.f;
      for (int j = 0; j < 23; ++j) o += sc[j] * qkv[j * 384 + 256 + hd * 16 + c];
      oab[r * 128 + hd * 16 + c] = (bf16_t)o;
    }
  }
  __syncthreads();

  // attended = o @ out_proj_w.T via WMMA
  for (int tid = wv; tid < 16; tid += 8) {
    int mt = tid / 8, nt = tid % 8;
    v8f acc = {};
#pragma unroll
    for (int k0 = 0; k0 < 128; k0 += 32) {
      v16bf a  = load_a_frag(oab + mt * 16 * 128 + k0, 128, lane);
      v16bf bb = load_b_frag(wout + nt * 16 * 128 + k0, 128, lane);
      acc = wmma_bf16(a, bb, acc);
    }
    store_d_tile(attm + mt * 16 * 128 + nt * 16, 128, acc, lane);
  }
  __syncthreads();

  if (t < 128) {
    float s = 0.f;
    for (int j = 0; j < 23; ++j) s += attm[j * 128 + t];
    gvec[t] = bout[t] + s * (1.0f / 23.0f);
  }
  for (int i = t; i < 529; i += 256) out_attnavg[b * 529 + i] = aavg[i];
  __syncthreads();

  // impacts: ih = relu(g @ iw1[e].T), logits = ih @ iw2[e].T, softmax
  for (int i = t; i < 9 * 256; i += 256) {
    int e = i >> 8, u = i & 255;
    const float* wr = iw1 + (e * 256 + u) * 128;
    float s = ib1[i];
    for (int d = 0; d < 128; ++d) s += gvec[d] * wr[d];
    ihs[i] = s > 0.f ? s : 0.f;
  }
  __syncthreads();
  for (int i = t; i < 207; i += 256) {
    int e = i / 23;
    const float* wr = iw2 + i * 256;
    const float* hr = ihs + e * 256;
    float s = ib2[i];
    for (int d = 0; d < 256; ++d) s += hr[d] * wr[d];
    lgts[i] = s;
  }
  __syncthreads();
  if (t < 9) {
    float mx = -1e30f;
    for (int f = 0; f < 23; ++f) { float v = lgts[t * 23 + f]; mx = v > mx ? v : mx; }
    float den = 0.f, ex[23];
    for (int f = 0; f < 23; ++f) { ex[f] = __expf(lgts[t * 23 + f] - mx); den += ex[f]; }
    float inv = 1.f / den;
    for (int f = 0; f < 23; ++f) out_impacts[b * 207 + t * 23 + f] = ex[f] * inv;
  }
}

// ---------------- one LSTM time step (both layers use this) ----------------
// grid (32 batch-tiles, 16 hidden-tiles), 4 waves = one wave per gate {i,f,g,o}.
// Activation tiles are async-staged global->LDS (contiguous 8KB rows), so the
// 4 waves share one copy instead of 4x redundant global reads.
__global__ __launch_bounds__(128) void lstm_step_kernel(
    const bf16_t* __restrict__ hprev, const bf16_t* __restrict__ whh,
    const bf16_t* __restrict__ xin,  const bf16_t* __restrict__ wih,
    const float* __restrict__ mcol,  const float* __restrict__ wsum,
    const float* __restrict__ bsum,
    float* __restrict__ cstate, bf16_t* __restrict__ hnext,
    float* __restrict__ hout, int hout_stride) {
  __shared__ __align__(16) bf16_t hs[16 * 256];  // 8KB h_prev tile
  __shared__ __align__(16) bf16_t xs[16 * 256];  // 8KB x tile (layer 1)
  __shared__ float gt[4][16][16];
  const int t = threadIdx.x, lane = t & 31, wv = t >> 5;
  const int m0 = blockIdx.x * 16;   // batch rows
  const int j0 = blockIdx.y * 16;   // hidden cols
  const int n0 = wv * 256 + j0;     // row in (4H x H) weight

  // async-stage the contiguous 16x256 bf16 tiles (512 x 16B chunks)
  {
    const char* hsrc = (const char*)(hprev + m0 * 256);
    unsigned hdst = lds_off_of(hs);
#pragma unroll
    for (int j = 0; j < 4; ++j) {
      int i = t + 128 * j;
      async_g2l_b128(hdst + i * 16, hsrc + i * 16);
    }
    if (xin) {
      const char* xsrc = (const char*)(xin + m0 * 256);
      unsigned xdst = lds_off_of(xs);
#pragma unroll
      for (int j = 0; j < 4; ++j) {
        int i = t + 128 * j;
        async_g2l_b128(xdst + i * 16, xsrc + i * 16);
      }
    }
  }
  wait_async0();
  __syncthreads();

  v8f acc = {};
#pragma unroll
  for (int k0 = 0; k0 < 256; k0 += 32) {
    v16bf a  = load_a_frag(hs + k0, 256, lane);
    v16bf bb = load_b_frag(whh + n0 * 256 + k0, 256, lane);
    acc = wmma_bf16(a, bb, acc);
  }
  if (xin) {  // layer 1: x @ w_ih.T (layer 0 uses rank-1 m*wsum instead)
#pragma unroll
    for (int k0 = 0; k0 < 256; k0 += 32) {
      v16bf a  = load_a_frag(xs + k0, 256, lane);
      v16bf bb = load_b_frag(wih + n0 * 256 + k0, 256, lane);
      acc = wmma_bf16(a, bb, acc);
    }
  }
  store_d_tile(&gt[wv][0][0], 16, acc, lane);
  __syncthreads();

  for (int idx = t; idx < 256; idx += 128) {
    int r = idx >> 4, cc = idx & 15;
    int bi = m0 + r, j = j0 + cc;
    float m = mcol ? mcol[bi * 40] : 0.0f;
    float gi = gt[0][r][cc] + bsum[j]       + (mcol ? m * wsum[j]       : 0.f);
    float gf = gt[1][r][cc] + bsum[256 + j] + (mcol ? m * wsum[256 + j] : 0.f);
    float gg = gt[2][r][cc] + bsum[512 + j] + (mcol ? m * wsum[512 + j] : 0.f);
    float go = gt[3][r][cc] + bsum[768 + j] + (mcol ? m * wsum[768 + j] : 0.f);
    float cold = cstate[bi * 256 + j];
    float cn = sigmoidf_(gf) * cold + sigmoidf_(gi) * fast_tanh(gg);
    cstate[bi * 256 + j] = cn;
    float h = sigmoidf_(go) * fast_tanh(cn);
    hnext[bi * 256 + j] = (bf16_t)h;
    if (hout) hout[bi * hout_stride + j] = h;
  }
}

// ---------------- trend head ----------------
__global__ __launch_bounds__(256) void trends_kernel(
    const float* __restrict__ temporal, const bf16_t* __restrict__ tw1bf,
    const float* __restrict__ tb1, const float* __restrict__ tw2,
    const float* __restrict__ tb2, float* __restrict__ out_trends) {
  __shared__ __align__(32) bf16_t tl[16 * 256];  // temporal[:, -1] tile (bf16)
  __shared__ float tt[16 * 512];
  const int t = threadIdx.x, lane = t & 31, wv = t >> 5;
  const int b0 = blockIdx.x * 16;

  for (int i = t; i < 16 * 256; i += 256) {
    int r = i >> 8, c = i & 255;
    tl[i] = (bf16_t)temporal[((b0 + r) * 40 + 39) * 256 + c];
  }
  __syncthreads();

  for (int nt = wv * 4; nt < wv * 4 + 4; ++nt) {  // 32 col-tiles / 8 waves
    v8f acc = {};
#pragma unroll
    for (int k0 = 0; k0 < 256; k0 += 32) {
      v16bf a  = load_a_frag(tl + k0, 256, lane);
      v16bf bb = load_b_frag(tw1bf + nt * 16 * 256 + k0, 256, lane);
      acc = wmma_bf16(a, bb, acc);
    }
    store_d_tile(tt + nt * 16, 512, acc, lane);
  }
  __syncthreads();
  for (int i = t; i < 16 * 512; i += 256) {
    float v = tt[i] + tb1[i & 511];
    tt[i] = v > 0.f ? v : 0.f;
  }
  __syncthreads();
  for (int i = t; i < 16 * 23; i += 256) {
    int r = i / 23, c = i % 23;
    const float* wr = tw2 + c * 512;
    const float* hr = tt + r * 512;
    float s = tb2[c];
    for (int d = 0; d < 512; ++d) s += hr[d] * wr[d];
    out_trends[(b0 + r) * 23 + c] = fast_tanh(s);
  }
}

static inline int cdiv(int a, int b) { return (a + b - 1) / b; }

extern "C" void kernel_launch(void* const* d_in, const int* in_sizes, int n_in,
                              void* d_out, int out_size, void* d_ws, size_t ws_size,
                              hipStream_t stream) {
  (void)in_sizes; (void)n_in; (void)out_size; (void)ws_size;
  const float* factors   = (const float*)d_in[0];
  const float* few1      = (const float*)d_in[1];
  const float* feb1      = (const float*)d_in[2];
  const float* few2      = (const float*)d_in[3];
  const float* feb2      = (const float*)d_in[4];
  const float* few3      = (const float*)d_in[5];
  const float* feb3      = (const float*)d_in[6];
  const float* in_proj_w = (const float*)d_in[7];
  const float* in_proj_b = (const float*)d_in[8];
  const float* out_proj_w= (const float*)d_in[9];
  const float* out_proj_b= (const float*)d_in[10];
  const float* iw1       = (const float*)d_in[11];
  const float* ib1       = (const float*)d_in[12];
  const float* iw2       = (const float*)d_in[13];
  const float* ib2       = (const float*)d_in[14];
  const float* w_ih0     = (const float*)d_in[15];
  const float* w_hh0     = (const float*)d_in[16];
  const float* b_ih0     = (const float*)d_in[17];
  const float* b_hh0     = (const float*)d_in[18];
  const float* w_ih1     = (const float*)d_in[19];
  const float* w_hh1     = (const float*)d_in[20];
  const float* b_ih1     = (const float*)d_in[21];
  const float* b_hh1     = (const float*)d_in[22];
  const float* tw1       = (const float*)d_in[23];
  const float* tb1       = (const float*)d_in[24];
  const float* tw2       = (const float*)d_in[25];
  const float* tb2       = (const float*)d_in[26];

  float* out          = (float*)d_out;
  float* out_impacts  = out;               // 512*9*23   = 105984
  float* out_attnavg  = out + 105984;      // 512*23*23  = 270848
  float* out_trends   = out + 376832;      // 512*23     = 11776
  float* out_temporal = out + 388608;      // 512*40*256 = 5242880

  char* ws = (char*)d_ws;
  size_t off = 0;
  auto take = [&](size_t bytes) {
    char* p = ws + off;
    off = (off + bytes + 255) & ~(size_t)255;
    return p;
  };
  bf16_t* wqkv_bf = (bf16_t*)take(384 * 128 * 2);
  bf16_t* wout_bf = (bf16_t*)take(128 * 128 * 2);
  bf16_t* whh0_bf = (bf16_t*)take(1024 * 256 * 2);
  bf16_t* wih1_bf = (bf16_t*)take(1024 * 256 * 2);
  bf16_t* whh1_bf = (bf16_t*)take(1024 * 256 * 2);
  bf16_t* tw1_bf  = (bf16_t*)take(512 * 256 * 2);
  float*  wsum0   = (float*)take(1024 * 4);
  float*  bsum0   = (float*)take(1024 * 4);
  float*  bsum1   = (float*)take(1024 * 4);
  float*  Mmean   = (float*)take(512 * 40 * 4);
  float*  c0      = (float*)take(512 * 256 * 4);
  float*  c1      = (float*)take(512 * 256 * 4);
  bf16_t* h0bf[2] = {(bf16_t*)take(512 * 256 * 2), (bf16_t*)take(512 * 256 * 2)};
  bf16_t* h1bf[2] = {(bf16_t*)take(512 * 256 * 2), (bf16_t*)take(512 * 256 * 2)};

  // ---- prep: bf16 weight copies, fused biases, rank-1 helpers, zero state ----
  f2bf_kernel<<<cdiv(384 * 128, 256), 256, 0, stream>>>(wqkv_bf, in_proj_w, 384 * 128);
  f2bf_kernel<<<cdiv(128 * 128, 256), 256, 0, stream>>>(wout_bf, out_proj_w, 128 * 128);
  f2bf_kernel<<<cdiv(1024 * 256, 256), 256, 0, stream>>>(whh0_bf, w_hh0, 1024 * 256);
  f2bf_kernel<<<cdiv(1024 * 256, 256), 256, 0, stream>>>(wih1_bf, w_ih1, 1024 * 256);
  f2bf_kernel<<<cdiv(1024 * 256, 256), 256, 0, stream>>>(whh1_bf, w_hh1, 1024 * 256);
  f2bf_kernel<<<cdiv(512 * 256, 256), 256, 0, stream>>>(tw1_bf, tw1, 512 * 256);
  rowsum23_kernel<<<4, 256, 0, stream>>>(wsum0, w_ih0, 1024);
  addv_kernel<<<4, 256, 0, stream>>>(bsum0, b_ih0, b_hh0, 1024);
  addv_kernel<<<4, 256, 0, stream>>>(bsum1, b_ih1, b_hh1, 1024);
  means_kernel<<<cdiv(512 * 40, 256), 256, 0, stream>>>(Mmean, factors, 512 * 40);
  zero_kernel<<<cdiv(512 * 256, 256), 256, 0, stream>>>(c0, 512 * 256);
  zero_kernel<<<cdiv(512 * 256, 256), 256, 0, stream>>>(c1, 512 * 256);
  zero_kernel<<<cdiv(512 * 128, 256), 256, 0, stream>>>((float*)h0bf[0], 512 * 128);
  zero_kernel<<<cdiv(512 * 128, 256), 256, 0, stream>>>((float*)h1bf[0], 512 * 128);

  // ---- encoder + attention + impacts ----
  attn_kernel<<<512, 256, 0, stream>>>(
      factors, few1, feb1, few2, feb2, few3, feb3,
      wqkv_bf, in_proj_b, wout_bf, out_proj_b,
      iw1, ib1, iw2, ib2, out_impacts, out_attnavg);

  // ---- two-layer LSTM, interleaved per step (layer1 consumes layer0's h directly) ----
  for (int s = 0; s < 40; ++s) {
    lstm_step_kernel<<<dim3(32, 16), 128, 0, stream>>>(
        h0bf[s & 1], whh0_bf, nullptr, nullptr,
        Mmean + s, wsum0, bsum0, c0, h0bf[(s + 1) & 1], nullptr, 0);
    lstm_step_kernel<<<dim3(32, 16), 128, 0, stream>>>(
        h1bf[s & 1], whh1_bf, h0bf[(s + 1) & 1], wih1_bf,
        nullptr, nullptr, bsum1, c1, h1bf[(s + 1) & 1],
        out_temporal + s * 256, 40 * 256);
  }

  // ---- trend head ----
  trends_kernel<<<32, 256, 0, stream>>>(out_temporal, tw1_bf, tb1, tw2, tb2, out_trends);
}
